// TestLSTMCell_54176717471880
// MI455X (gfx1250) — matrix operand order repaired and checked by reading, not compiled, not run on hardware
//
#include <hip/hip_runtime.h>
#include <hip/hip_bf16.h>

// ---------------------------------------------------------------------------
// Types for CDNA5 WMMA
// ---------------------------------------------------------------------------
typedef __attribute__((ext_vector_type(16))) __bf16 v16bf;
typedef __attribute__((ext_vector_type(8)))  __bf16 v8bf;
typedef __attribute__((ext_vector_type(4)))  __bf16 v4bf;
typedef __attribute__((ext_vector_type(8)))  float  v8f;

#define BATCH 4096
#define HID   1024
#define KDIM  1024

// ---------------------------------------------------------------------------
// Fast branch-free activations (v_exp_f32 + v_rcp_f32)
// ---------------------------------------------------------------------------
__device__ __forceinline__ float fast_sigmoid(float x) {
    return __builtin_amdgcn_rcpf(1.0f + __expf(-x));
}
__device__ __forceinline__ float fast_tanh(float x) {
    // tanh(x) = 1 - 2/(exp(2x)+1); exp->inf gives 1, exp->0 gives -1
    return 1.0f - 2.0f * __builtin_amdgcn_rcpf(__expf(2.0f * x) + 1.0f);
}

__device__ __forceinline__ v16bf cat16(v8bf lo, v8bf hi) {
    v16bf r;
#pragma unroll
    for (int i = 0; i < 8; ++i) { r[i] = lo[i]; r[i + 8] = hi[i]; }
    return r;
}

// A-fragment (16x32 bf16, M x K), row-major source with row stride KDIM.
// Lanes 0-15: row M=lane, K = {k0..k0+7} U {k0+16..k0+23}
// Lanes 16-31: row M=lane-16, K = {k0+8..k0+15} U {k0+24..k0+31}
__device__ __forceinline__ v16bf load_frag_a(const __bf16* __restrict__ A,
                                             int row0, int k0, int lane) {
    int r  = row0 + (lane & 15);
    int kb = k0 + ((lane >> 4) << 3);
    const __bf16* p = A + (size_t)r * KDIM + kb;
    v8bf lo = *(const v8bf*)p;
    v8bf hi = *(const v8bf*)(p + 16);
    return cat16(lo, hi);
}

// B-fragment (32x16 bf16, K x N) with B[k,n] = W[n,k]; W row-major, stride KDIM.
// Lanes 0-15: col N=lane, K = {k0..k0+7} U {k0+16..k0+23}; lanes 16-31: K = {k0+8..k0+15} U {k0+24..k0+31}
__device__ __forceinline__ v16bf load_frag_b(const __bf16* __restrict__ W,
                                             int col0, int k0, int lane) {
    int n  = col0 + (lane & 15);
    int kb = k0 + ((lane >> 4) << 3);
    const __bf16* p = W + (size_t)n * KDIM + kb;
    v8bf lo = *(const v8bf*)p;
    v8bf hi = *(const v8bf*)(p + 16);
    return cat16(lo, hi);
}

__device__ __forceinline__ v8f wmma_bf16(v16bf a, v16bf b, v8f c) {
    return __builtin_amdgcn_wmma_f32_16x16x32_bf16(
        false, a, false, b, (short)0, c, false, false);
}

__device__ __forceinline__ v8f vzero() {
    v8f z = {0.f, 0.f, 0.f, 0.f, 0.f, 0.f, 0.f, 0.f};
    return z;
}

// Block tiling: 256 threads = 8 waves; block tile 128x128; wave tile 64x32
// (4 x 2 WMMA tiles of 16x16).  Grid: (HID/128, BATCH/128).
struct TileCoords { int lane, rowBase, colBase; };
__device__ __forceinline__ TileCoords tile_coords() {
    TileCoords t;
    t.lane = threadIdx.x & 31;
    int wave = threadIdx.x >> 5;
    t.rowBase = blockIdx.y * 128 + (wave >> 2) * 64;
    t.colBase = blockIdx.x * 128 + (wave & 3) * 32;
    return t;
}

// Fragment-group loaders / MMA steps --------------------------------------
__device__ __forceinline__ void load_a4(const __bf16* __restrict__ A, int rowBase,
                                        int k, int lane, v16bf a[4]) {
#pragma unroll
    for (int i = 0; i < 4; ++i) a[i] = load_frag_a(A, rowBase + i * 16, k, lane);
}
__device__ __forceinline__ void load_b2(const __bf16* __restrict__ W, int colBase,
                                        int k, int lane, v16bf b[2]) {
#pragma unroll
    for (int j = 0; j < 2; ++j) b[j] = load_frag_b(W, colBase + j * 16, k, lane);
}
__device__ __forceinline__ void mma_42(v8f acc[4][2], const v16bf a[4], const v16bf b[2]) {
#pragma unroll
    for (int i = 0; i < 4; ++i)
#pragma unroll
        for (int j = 0; j < 2; ++j) acc[i][j] = wmma_bf16(a[i], b[j], acc[i][j]);
}

// ---------------------------------------------------------------------------
// fp32 -> bf16 conversion (vectorized x4)
// ---------------------------------------------------------------------------
__global__ __launch_bounds__(256) void cvt_f32_bf16(const float* __restrict__ in,
                                                    __bf16* __restrict__ out, int n4) {
    int i = blockIdx.x * 256 + threadIdx.x;
    if (i < n4) {
        float4 v = ((const float4*)in)[i];
        v4bf o;
        o[0] = (__bf16)v.x; o[1] = (__bf16)v.y;
        o[2] = (__bf16)v.z; o[3] = (__bf16)v.w;
        ((v4bf*)out)[i] = o;
    }
}

// ---------------------------------------------------------------------------
// Stage 1: t = tanh(x @ W_xt^T + b_xt) + hx ; write t as f32 and bf16
// Software-pipelined K loop (double-buffered fragments).
// ---------------------------------------------------------------------------
__global__ __launch_bounds__(256, 1) void g1_kernel(
    const __bf16* __restrict__ xb, const __bf16* __restrict__ w,
    const float* __restrict__ bias, const float* __restrict__ hx,
    float* __restrict__ tf32, __bf16* __restrict__ tb) {

    TileCoords tc = tile_coords();
    v8f acc[4][2];
#pragma unroll
    for (int i = 0; i < 4; ++i)
#pragma unroll
        for (int j = 0; j < 2; ++j) acc[i][j] = vzero();

    v16bf a0[4], b0[2], a1[4], b1[2];
    load_a4(xb, tc.rowBase, 0, tc.lane, a0);
    load_b2(w,  tc.colBase, 0, tc.lane, b0);

    int k = 0;
    for (; k + 64 < KDIM; k += 64) {
        load_a4(xb, tc.rowBase, k + 32, tc.lane, a1);
        load_b2(w,  tc.colBase, k + 32, tc.lane, b1);
        mma_42(acc, a0, b0);
        load_a4(xb, tc.rowBase, k + 64, tc.lane, a0);
        load_b2(w,  tc.colBase, k + 64, tc.lane, b0);
        mma_42(acc, a1, b1);
    }
    load_a4(xb, tc.rowBase, k + 32, tc.lane, a1);
    load_b2(w,  tc.colBase, k + 32, tc.lane, b1);
    mma_42(acc, a0, b0);
    mma_42(acc, a1, b1);

    int mHi = (tc.lane >> 4) << 3;
    int nC  = tc.lane & 15;
#pragma unroll
    for (int i = 0; i < 4; ++i)
#pragma unroll
        for (int j = 0; j < 2; ++j) {
            int col = tc.colBase + j * 16 + nC;
            float bv = bias[col];
#pragma unroll
            for (int v = 0; v < 8; ++v) {
                int row = tc.rowBase + i * 16 + mHi + v;
                size_t idx = (size_t)row * HID + col;
                float t = fast_tanh(acc[i][j][v] + bv) + hx[idx];
                tf32[idx] = t;
                tb[idx]   = (__bf16)t;
            }
        }
}

// ---------------------------------------------------------------------------
// Stage 2 (fused): accF = t@W_tf^T + cx@W_cf^T ; accU = t@W_tu^T + cx@W_cu^T ;
//                  accH = t@W_th^T.
// Epilogue: f = sig(accF+b), u = sig(accU+b)*t, cy = tanh(f*cx+u),
//           hpre = accH + b_th.
// ---------------------------------------------------------------------------
__global__ __launch_bounds__(256, 1) void g2_kernel(
    const __bf16* __restrict__ tb, const __bf16* __restrict__ cxb,
    const __bf16* __restrict__ wtf, const __bf16* __restrict__ wcf,
    const __bf16* __restrict__ wtu, const __bf16* __restrict__ wcu,
    const __bf16* __restrict__ wth,
    const float* __restrict__ btf, const float* __restrict__ bcf,
    const float* __restrict__ btu, const float* __restrict__ bcu,
    const float* __restrict__ bth,
    const float* __restrict__ tf32, const float* __restrict__ cx,
    float* __restrict__ cyOut, __bf16* __restrict__ cyb,
    float* __restrict__ hpre) {

    TileCoords tc = tile_coords();
    v8f accF[4][2], accU[4][2], accH[4][2];
#pragma unroll
    for (int i = 0; i < 4; ++i)
#pragma unroll
        for (int j = 0; j < 2; ++j) {
            accF[i][j] = vzero(); accU[i][j] = vzero(); accH[i][j] = vzero();
        }

    // ---- Pass 1: A = t ; weights W_tf, W_tu, W_th (pipelined) ----
    {
        v16bf a0[4], bF0[2], bU0[2], bH0[2];
        v16bf a1[4], bF1[2], bU1[2], bH1[2];
        load_a4(tb,  tc.rowBase, 0, tc.lane, a0);
        load_b2(wtf, tc.colBase, 0, tc.lane, bF0);
        load_b2(wtu, tc.colBase, 0, tc.lane, bU0);
        load_b2(wth, tc.colBase, 0, tc.lane, bH0);
        int k = 0;
        for (; k + 64 < KDIM; k += 64) {
            load_a4(tb,  tc.rowBase, k + 32, tc.lane, a1);
            load_b2(wtf, tc.colBase, k + 32, tc.lane, bF1);
            load_b2(wtu, tc.colBase, k + 32, tc.lane, bU1);
            load_b2(wth, tc.colBase, k + 32, tc.lane, bH1);
            mma_42(accF, a0, bF0); mma_42(accU, a0, bU0); mma_42(accH, a0, bH0);
            load_a4(tb,  tc.rowBase, k + 64, tc.lane, a0);
            load_b2(wtf, tc.colBase, k + 64, tc.lane, bF0);
            load_b2(wtu, tc.colBase, k + 64, tc.lane, bU0);
            load_b2(wth, tc.colBase, k + 64, tc.lane, bH0);
            mma_42(accF, a1, bF1); mma_42(accU, a1, bU1); mma_42(accH, a1, bH1);
        }
        load_a4(tb,  tc.rowBase, k + 32, tc.lane, a1);
        load_b2(wtf, tc.colBase, k + 32, tc.lane, bF1);
        load_b2(wtu, tc.colBase, k + 32, tc.lane, bU1);
        load_b2(wth, tc.colBase, k + 32, tc.lane, bH1);
        mma_42(accF, a0, bF0); mma_42(accU, a0, bU0); mma_42(accH, a0, bH0);
        mma_42(accF, a1, bF1); mma_42(accU, a1, bU1); mma_42(accH, a1, bH1);
    }

    // ---- Pass 2: A = cx ; weights W_cf, W_cu (pipelined) ----
    {
        v16bf a0[4], bF0[2], bU0[2];
        v16bf a1[4], bF1[2], bU1[2];
        load_a4(cxb, tc.rowBase, 0, tc.lane, a0);
        load_b2(wcf, tc.colBase, 0, tc.lane, bF0);
        load_b2(wcu, tc.colBase, 0, tc.lane, bU0);
        int k = 0;
        for (; k + 64 < KDIM; k += 64) {
            load_a4(cxb, tc.rowBase, k + 32, tc.lane, a1);
            load_b2(wcf, tc.colBase, k + 32, tc.lane, bF1);
            load_b2(wcu, tc.colBase, k + 32, tc.lane, bU1);
            mma_42(accF, a0, bF0); mma_42(accU, a0, bU0);
            load_a4(cxb, tc.rowBase, k + 64, tc.lane, a0);
            load_b2(wcf, tc.colBase, k + 64, tc.lane, bF0);
            load_b2(wcu, tc.colBase, k + 64, tc.lane, bU0);
            mma_42(accF, a1, bF1); mma_42(accU, a1, bU1);
        }
        load_a4(cxb, tc.rowBase, k + 32, tc.lane, a1);
        load_b2(wcf, tc.colBase, k + 32, tc.lane, bF1);
        load_b2(wcu, tc.colBase, k + 32, tc.lane, bU1);
        mma_42(accF, a0, bF0); mma_42(accU, a0, bU0);
        mma_42(accF, a1, bF1); mma_42(accU, a1, bU1);
    }

    int mHi = (tc.lane >> 4) << 3;
    int nC  = tc.lane & 15;
#pragma unroll
    for (int i = 0; i < 4; ++i)
#pragma unroll
        for (int j = 0; j < 2; ++j) {
            int col = tc.colBase + j * 16 + nC;
            float bF = btf[col] + bcf[col];
            float bU = btu[col] + bcu[col];
            float bH = bth[col];
#pragma unroll
            for (int v = 0; v < 8; ++v) {
                int row = tc.rowBase + i * 16 + mHi + v;
                size_t idx = (size_t)row * HID + col;
                float f  = fast_sigmoid(accF[i][j][v] + bF);
                float u  = fast_sigmoid(accU[i][j][v] + bU) * tf32[idx];
                float cy = fast_tanh(f * cx[idx] + u);
                cyOut[idx] = cy;
                cyb[idx]   = (__bf16)cy;
                hpre[idx]  = accH[i][j][v] + bH;
            }
        }
}

// ---------------------------------------------------------------------------
// Stage 3: hy = tanh( sigmoid(hpre + cy@W_ch^T + b_ch) * cy )
// ---------------------------------------------------------------------------
__global__ __launch_bounds__(256, 1) void g3_kernel(
    const __bf16* __restrict__ cyb, const __bf16* __restrict__ wch,
    const float* __restrict__ bch, const float* __restrict__ hpre,
    const float* __restrict__ cyf, float* __restrict__ hy) {

    TileCoords tc = tile_coords();
    v8f acc[4][2];
#pragma unroll
    for (int i = 0; i < 4; ++i)
#pragma unroll
        for (int j = 0; j < 2; ++j) acc[i][j] = vzero();

    v16bf a0[4], b0[2], a1[4], b1[2];
    load_a4(cyb, tc.rowBase, 0, tc.lane, a0);
    load_b2(wch, tc.colBase, 0, tc.lane, b0);
    int k = 0;
    for (; k + 64 < KDIM; k += 64) {
        load_a4(cyb, tc.rowBase, k + 32, tc.lane, a1);
        load_b2(wch, tc.colBase, k + 32, tc.lane, b1);
        mma_42(acc, a0, b0);
        load_a4(cyb, tc.rowBase, k + 64, tc.lane, a0);
        load_b2(wch, tc.colBase, k + 64, tc.lane, b0);
        mma_42(acc, a1, b1);
    }
    load_a4(cyb, tc.rowBase, k + 32, tc.lane, a1);
    load_b2(wch, tc.colBase, k + 32, tc.lane, b1);
    mma_42(acc, a0, b0);
    mma_42(acc, a1, b1);

    int mHi = (tc.lane >> 4) << 3;
    int nC  = tc.lane & 15;
#pragma unroll
    for (int i = 0; i < 4; ++i)
#pragma unroll
        for (int j = 0; j < 2; ++j) {
            int col = tc.colBase + j * 16 + nC;
            float bv = bch[col];
#pragma unroll
            for (int v = 0; v < 8; ++v) {
                int row = tc.rowBase + i * 16 + mHi + v;
                size_t idx = (size_t)row * HID + col;
                float s = fast_sigmoid(acc[i][j][v] + bv + hpre[idx]);
                hy[idx] = fast_tanh(s * cyf[idx]);
            }
        }
}

// ---------------------------------------------------------------------------
// Host-side launcher
// ---------------------------------------------------------------------------
extern "C" void kernel_launch(void* const* d_in, const int* in_sizes, int n_in,
                              void* d_out, int out_size, void* d_ws, size_t ws_size,
                              hipStream_t stream) {
    (void)in_sizes; (void)n_in; (void)out_size; (void)ws_size;

    const float* x    = (const float*)d_in[0];
    const float* hx   = (const float*)d_in[1];
    const float* cx   = (const float*)d_in[2];
    const float* W[7] = {(const float*)d_in[3], (const float*)d_in[4],
                         (const float*)d_in[5], (const float*)d_in[6],
                         (const float*)d_in[7], (const float*)d_in[8],
                         (const float*)d_in[9]};
    const float* b_xt = (const float*)d_in[10];
    const float* b_tf = (const float*)d_in[11];
    const float* b_cf = (const float*)d_in[12];
    const float* b_tu = (const float*)d_in[13];
    const float* b_cu = (const float*)d_in[14];
    const float* b_th = (const float*)d_in[15];
    const float* b_ch = (const float*)d_in[16];

    float* hy_out = (float*)d_out;
    float* cy_out = (float*)d_out + (size_t)BATCH * HID;

    const size_t ACT_BF = (size_t)BATCH * HID * sizeof(__bf16);
    const size_t W_BF   = (size_t)HID * KDIM * sizeof(__bf16);
    const size_t ACT_F  = (size_t)BATCH * HID * sizeof(float);
    char* ws = (char*)d_ws;
    __bf16* xb   = (__bf16*)(ws);                 ws += ACT_BF;
    __bf16* cxb  = (__bf16*)(ws);                 ws += ACT_BF;
    __bf16* tbuf = (__bf16*)(ws);                 ws += ACT_BF;
    __bf16* cyb  = (__bf16*)(ws);                 ws += ACT_BF;
    __bf16* Wb[7];
    for (int i = 0; i < 7; ++i) { Wb[i] = (__bf16*)ws; ws += W_BF; }
    float* tf32  = (float*)(ws);                  ws += ACT_F;
    float* hpre  = (float*)(ws);                  ws += ACT_F;

    {
        int n4 = (BATCH * HID) / 4;
        int g  = (n4 + 255) / 256;
        cvt_f32_bf16<<<g, 256, 0, stream>>>(x,  xb,  n4);
        cvt_f32_bf16<<<g, 256, 0, stream>>>(cx, cxb, n4);
        int n4w = (HID * KDIM) / 4;
        int gw  = (n4w + 255) / 256;
        for (int i = 0; i < 7; ++i)
            cvt_f32_bf16<<<gw, 256, 0, stream>>>(W[i], Wb[i], n4w);
    }

    dim3 grid(HID / 128, BATCH / 128);
    dim3 block(256);

    g1_kernel<<<grid, block, 0, stream>>>(xb, Wb[0], b_xt, hx, tf32, tbuf);
    g2_kernel<<<grid, block, 0, stream>>>(tbuf, cxb,
                                          Wb[1], Wb[2], Wb[3], Wb[4], Wb[5],
                                          b_tf, b_cf, b_tu, b_cu, b_th,
                                          tf32, cx, cy_out, cyb, hpre);
    g3_kernel<<<grid, block, 0, stream>>>(cyb, Wb[6], b_ch, hpre, cy_out, hy_out);
}
